// TabNSA_74311524155774
// MI455X (gfx1250) — matrix operand order repaired and checked by reading, not compiled, not run on hardware
//
#include <hip/hip_runtime.h>
#include <hip/hip_bf16.h>

typedef __attribute__((ext_vector_type(16))) _Float16 v16h;
typedef __attribute__((ext_vector_type(8)))  float    v8f;

constexpr int B_ = 128, N_ = 128, D_ = 64, H_ = 8, DH_ = 64, HD_ = 512;
constexpr int CBS_ = 4, SBS_ = 4, NB_ = 32;
constexpr float NEGF = -1e30f;

__device__ __forceinline__ float gelu_f(float x) {
    return 0.5f * x * (1.0f + erff(x * 0.7071067811865475f));
}

__device__ __forceinline__ float wred32(float p) {
#pragma unroll
    for (int o = 16; o >= 1; o >>= 1) p += __shfl_xor(p, o, 32);
    return p;
}

// ---------------------------------------------------------------------------
// f32 -> f16 convert (weights)
// ---------------------------------------------------------------------------
__global__ __launch_bounds__(256) void cvt16_kernel(const float* __restrict__ s,
                                                    _Float16* __restrict__ d, int n) {
    int i = blockIdx.x * 256 + threadIdx.x;
    if (i < n) d[i] = (_Float16)s[i];
}

// ---------------------------------------------------------------------------
// Embed + RMSNorm + LN1 (for token mixer, stored transposed (B,D,N) as f16)
// One wave (32 lanes) per (b,n) row; each lane owns dims {lane, lane+32}.
// ---------------------------------------------------------------------------
__global__ __launch_bounds__(32)
void embed_kernel(const float* __restrict__ x, const float* __restrict__ W_emb,
                  const float* __restrict__ b_emb, const float* __restrict__ rms_scale,
                  const float* __restrict__ ln1_g, const float* __restrict__ ln1_b,
                  float* __restrict__ e, float* __restrict__ hbuf,
                  _Float16* __restrict__ h16, _Float16* __restrict__ t16) {
    int row = blockIdx.x;           // b*N + n
    int lane = threadIdx.x;
    int d0 = lane, d1 = lane + 32;
    float xv = x[row];
    float e0 = xv * W_emb[d0] + b_emb[d0];
    float e1 = xv * W_emb[d1] + b_emb[d1];
    size_t off = (size_t)row * D_;
    e[off + d0] = e0; e[off + d1] = e1;
    // RMS norm (eps 1e-6)
    float ms = wred32(e0 * e0 + e1 * e1) * (1.0f / D_);
    float ir = rsqrtf(ms + 1e-6f);
    float h0 = e0 * ir * rms_scale[d0];
    float h1 = e1 * ir * rms_scale[d1];
    hbuf[off + d0] = h0; hbuf[off + d1] = h1;
    h16[off + d0] = (_Float16)h0; h16[off + d1] = (_Float16)h1;
    // LayerNorm (eps 1e-5) -> transposed f16 for token mixer
    float mu = wred32(e0 + e1) * (1.0f / D_);
    float dv0 = e0 - mu, dv1 = e1 - mu;
    float var = wred32(dv0 * dv0 + dv1 * dv1) * (1.0f / D_);
    float isd = rsqrtf(var + 1e-5f);
    float l0 = dv0 * isd * ln1_g[d0] + ln1_b[d0];
    float l1 = dv1 * isd * ln1_g[d1] + ln1_b[d1];
    int b = row >> 7, n = row & (N_ - 1);
    t16[((size_t)(b * D_ + d0)) * N_ + n] = (_Float16)l0;
    t16[((size_t)(b * D_ + d1)) * N_ + n] = (_Float16)l1;
}

// ---------------------------------------------------------------------------
// LayerNorm over D=64 -> f16 rows (channel mixer input)
// ---------------------------------------------------------------------------
__global__ __launch_bounds__(32)
void ln_kernel(const float* __restrict__ in, const float* __restrict__ g,
               const float* __restrict__ bta, _Float16* __restrict__ out16) {
    int row = blockIdx.x;
    int lane = threadIdx.x;
    size_t off = (size_t)row * D_;
    float a0 = in[off + lane], a1 = in[off + lane + 32];
    float mu = wred32(a0 + a1) * (1.0f / D_);
    float d0 = a0 - mu, d1 = a1 - mu;
    float var = wred32(d0 * d0 + d1 * d1) * (1.0f / D_);
    float isd = rsqrtf(var + 1e-5f);
    out16[off + lane]      = (_Float16)(d0 * isd * g[lane] + bta[lane]);
    out16[off + lane + 32] = (_Float16)(d1 * isd * g[lane + 32] + bta[lane + 32]);
}

// ---------------------------------------------------------------------------
// WMMA GEMM: C[M,Nc] = act(A16[M,K] @ B16[K,Nc] + bias), optional f16 mirror.
// Each wave computes a 16x64 strip: one A fragment per K-step feeds 4
// independent v_wmma_f32_16x16x32_f16 (4 N-tiles) -> 4x less A traffic and
// no WMMA->WMMA RAW hazards. global_prefetch_b8 pulls the next K-step.
// M mult of 16; Nc mult of 64; K mult of 32 (true for all GEMMs here).
// ---------------------------------------------------------------------------
template <int ACT, bool W16>
__global__ __launch_bounds__(256)
void gemm_wmma_kernel(const _Float16* __restrict__ A, const _Float16* __restrict__ Bw,
                      const float* __restrict__ bias, float* __restrict__ C,
                      _Float16* __restrict__ C16, int M, int Nc, int K) {
    int wid = blockIdx.x * 8 + (threadIdx.x >> 5);
    int nquad = Nc >> 6;                      // 64-wide column strips
    int tiles = (M >> 4) * nquad;
    if (wid >= tiles) return;                 // wave-uniform exit (EXEC stays all-1s)
    int tm = wid / nquad, tq = wid % nquad;
    int lane = threadIdx.x & 31;
    int la = lane & 15;
    int hi = lane >> 4;                       // 0: lanes 0-15, 1: lanes 16-31
    int koff = hi * 8;
    const _Float16* arow = A + (size_t)(tm * 16 + la) * K;
    v8f acc[4];
#pragma unroll
    for (int t = 0; t < 4; ++t) acc[t] = v8f{0.f, 0.f, 0.f, 0.f, 0.f, 0.f, 0.f, 0.f};
    for (int k0 = 0; k0 < K; k0 += 32) {
        v16h af;
#pragma unroll
        for (int j = 0; j < 8; ++j) {         // A 16x32: V0-3 -> K grp 0, V4-7 -> K grp 16
            int kk = k0 + ((j < 4) ? 0 : 16) + koff + 2 * (j & 3);
            af[2 * j]     = arow[kk];
            af[2 * j + 1] = arow[kk + 1];
        }
        const _Float16* bbase = Bw + (size_t)k0 * Nc + tq * 64 + la;  // lane = N column
        if (k0 + 32 < K) {                    // prefetch next K-step (gfx1250 path)
            __builtin_prefetch(arow + k0 + 32, 0, 1);
            __builtin_prefetch(bbase + (size_t)32 * Nc, 0, 1);
        }
#pragma unroll
        for (int t = 0; t < 4; ++t) {
            v16h bf;
#pragma unroll
            for (int j = 0; j < 16; ++j) bf[j] = bbase[(size_t)(koff + j + ((j < 8) ? 0 : 8)) * Nc + t * 16];       // elem = K (table order)
            acc[t] = __builtin_amdgcn_wmma_f32_16x16x32_f16(false, af, false, bf,
                                                            (short)0, acc[t], false, false);
        }
    }
    int mo = tm * 16 + hi * 8;
#pragma unroll
    for (int t = 0; t < 4; ++t) {
        int no = tq * 64 + t * 16 + la;
        float bv = bias ? bias[no] : 0.f;
#pragma unroll
        for (int r = 0; r < 8; ++r) {
            float val = acc[t][r] + bv;
            if (ACT == 1) val = gelu_f(val);
            size_t o = (size_t)(mo + r) * Nc + no;
            C[o] = val;
            if (W16) C16[o] = (_Float16)val;
        }
    }
}

// ---------------------------------------------------------------------------
// Gates: G[(b*N+n)*24 + comp*8 + h] = sigmoid(h @ Wg + bg)
// ---------------------------------------------------------------------------
__global__ __launch_bounds__(256)
void gate_kernel(const float* __restrict__ h, const float* __restrict__ Wg,
                 const float* __restrict__ bg, float* __restrict__ G) {
    int idx = blockIdx.x * 256 + threadIdx.x;
    if (idx >= B_ * N_ * 24) return;
    int c = idx % 24, row = idx / 24;
    const float* hr = h + (size_t)row * D_;
    float acc = bg[c];
    for (int d = 0; d < D_; ++d) acc += hr[d] * Wg[d * 24 + c];
    G[idx] = 1.f / (1.f + __expf(-acc));
}

// ---------------------------------------------------------------------------
// kb/vb = per-block K/V + positional, flattened to (B*H*NB, 256) f16
// ---------------------------------------------------------------------------
__global__ __launch_bounds__(256)
void kbvb_kernel(const float* __restrict__ K, const float* __restrict__ V,
                 const float* __restrict__ k_pos, const float* __restrict__ v_pos,
                 _Float16* __restrict__ kb16, _Float16* __restrict__ vb16) {
    int idx = blockIdx.x * 256 + threadIdx.x;
    if (idx >= B_ * H_ * NB_ * 256) return;
    int j = idx & 255, r = idx >> 8;          // r = (b*H+h)*NB + nb
    int nb = r & (NB_ - 1), bh = r >> 5;
    int h = bh & (H_ - 1), b = bh >> 3;
    int cb = j >> 6, dh = j & 63;
    int tok = nb * CBS_ + cb;
    size_t ko = ((size_t)(b * N_ + tok)) * HD_ + h * DH_ + dh;
    size_t po = (size_t)(h * CBS_ + cb) * DH_ + dh;
    kb16[idx] = (_Float16)(K[ko] + k_pos[po]);
    vb16[idx] = (_Float16)(V[ko] + v_pos[po]);
}

// ---------------------------------------------------------------------------
// Fused 3-branch attention + gating. One wave per (b,h,n); lane owns dims
// {lane, lane+32}. All branch decisions are wave-uniform (safe for shfl).
// ---------------------------------------------------------------------------
__global__ __launch_bounds__(128)
void attn_kernel(const float* __restrict__ Q, const float* __restrict__ K,
                 const float* __restrict__ V, const float* __restrict__ ck,
                 const float* __restrict__ cv, const float* __restrict__ mem_ck,
                 const float* __restrict__ mem_cv, const float* __restrict__ G,
                 _Float16* __restrict__ attn16) {
    int wid = blockIdx.x * 4 + (threadIdx.x >> 5);   // (b*H+h)*N + n
    int lane = threadIdx.x & 31;
    int n = wid & (N_ - 1);
    int bh = wid >> 7;
    int h = bh & (H_ - 1), b = bh >> 3;
    const float scale = 0.125f;                       // DH^-0.5

    size_t qoff = ((size_t)(b * N_ + n)) * HD_ + h * DH_;
    float q0 = Q[qoff + lane], q1 = Q[qoff + lane + 32];

    auto wdot = [&](const float* row) -> float {
        return wred32(q0 * row[lane] + q1 * row[lane + 32]);
    };

    // ---- compressed branch: scores vs [mem, 32 blocks], causal block mask
    float csv[33];
    csv[0] = wdot(mem_ck + h * DH_) * scale;
    const float* ckb = ck + (size_t)bh * NB_ * DH_;
    const float* cvb = cv + (size_t)bh * NB_ * DH_;
    for (int j = 1; j <= NB_; ++j) {
        float s = wdot(ckb + (j - 1) * DH_) * scale;
        csv[j] = (n >= j * CBS_ - 1) ? s : NEGF;      // pos >= (j)*4 - 1
    }
    float mx = csv[0];
    for (int j = 1; j < 33; ++j) mx = fmaxf(mx, csv[j]);
    float sum = 0.f, oc0 = 0.f, oc1 = 0.f;
    for (int j = 0; j < 33; ++j) {
        float w = __expf(csv[j] - mx);
        sum += w;
        const float* row = (j == 0) ? (mem_cv + h * DH_) : (cvb + (j - 1) * DH_);
        oc0 += w * row[lane];
        oc1 += w * row[lane + 32];
    }
    float inv = 1.f / sum; oc0 *= inv; oc1 *= inv;

    // ---- top-2 block selection over csv[1..32] (masked scores)
    float v1 = -3.4e38f, v2 = -3.4e38f; int i1 = 0, i2 = 0;
    for (int jb = 0; jb < NB_; ++jb) {
        float s = csv[jb + 1];
        if (s > v1)      { v2 = v1; i2 = i1; v1 = s; i1 = jb; }
        else if (s > v2) { v2 = s;  i2 = jb; }
    }
    int qblk = n >> 2;                                 // n / SBS
    bool ok1 = (v1 > NEGF * 0.5f) && (i1 != qblk);
    bool ok2 = (v2 > NEGF * 0.5f) && (i2 != qblk);

    // ---- fine branch: 8 selected tokens + 4 diagonal tokens
    float fsv[12]; int ftok[12];
    for (int s = 0; s < 12; ++s) {
        int tok; bool msk;
        if (s < 4)      { tok = i1 * SBS_ + s;        msk = ok1 && (tok <= n); }
        else if (s < 8) { tok = i2 * SBS_ + (s - 4);  msk = ok2 && (tok <= n); }
        else            { tok = qblk * SBS_ + (s - 8); msk = (tok <= n); }
        ftok[s] = tok;
        fsv[s] = msk ? wdot(K + ((size_t)(b * N_ + tok)) * HD_ + h * DH_) * scale : NEGF;
    }
    float mxf = fsv[0];
    for (int s = 1; s < 12; ++s) mxf = fmaxf(mxf, fsv[s]);
    float sumf = 0.f, of0 = 0.f, of1 = 0.f;
    for (int s = 0; s < 12; ++s) {
        float w = __expf(fsv[s] - mxf);
        sumf += w;
        const float* row = V + ((size_t)(b * N_ + ftok[s])) * HD_ + h * DH_;
        of0 += w * row[lane];
        of1 += w * row[lane + 32];
    }
    float invf = 1.f / sumf; of0 *= invf; of1 *= invf;

    // ---- window branch (WIN=2): tokens {n-1, n}
    int t0 = (n > 0) ? (n - 1) : 0;
    float ws0 = (n > 0) ? wdot(K + ((size_t)(b * N_ + t0)) * HD_ + h * DH_) * scale : NEGF;
    float ws1 = wdot(K + ((size_t)(b * N_ + n)) * HD_ + h * DH_) * scale;
    float mxw = fmaxf(ws0, ws1);
    float w0 = __expf(ws0 - mxw), w1 = __expf(ws1 - mxw);
    float invw = 1.f / (w0 + w1);
    const float* vr0 = V + ((size_t)(b * N_ + t0)) * HD_ + h * DH_;
    const float* vr1 = V + ((size_t)(b * N_ + n)) * HD_ + h * DH_;
    float ow0 = (w0 * vr0[lane]      + w1 * vr1[lane])      * invw;
    float ow1 = (w0 * vr0[lane + 32] + w1 * vr1[lane + 32]) * invw;

    // ---- gated merge, store f16 row-major (B*N, H*DH) for the Wo GEMM
    size_t goff = ((size_t)(b * N_ + n)) * 24;
    float g0 = G[goff + h], g1 = G[goff + 8 + h], g2 = G[goff + 16 + h];
    attn16[qoff + lane]      = (_Float16)(g0 * oc0 + g1 * of0 + g2 * ow0);
    attn16[qoff + lane + 32] = (_Float16)(g0 * oc1 + g1 * of1 + g2 * ow1);
}

// ---------------------------------------------------------------------------
// m = e + transpose(token-mixer output (B,D,N))
// ---------------------------------------------------------------------------
__global__ __launch_bounds__(256)
void addT_kernel(const float* __restrict__ e, const float* __restrict__ Vt,
                 float* __restrict__ m) {
    int idx = blockIdx.x * 256 + threadIdx.x;
    if (idx >= B_ * N_ * D_) return;
    int d = idx & 63, row = idx >> 6;
    int b = row >> 7, n = row & (N_ - 1);
    m[idx] = e[idx] + Vt[((size_t)(b * D_ + d)) * N_ + n];
}

// ---------------------------------------------------------------------------
// y[b,d] = mean_n (x1 + m + Qc)
// ---------------------------------------------------------------------------
__global__ __launch_bounds__(256)
void mean_kernel(const float* __restrict__ x1, const float* __restrict__ m,
                 const float* __restrict__ Qc, float* __restrict__ y) {
    int idx = blockIdx.x * 256 + threadIdx.x;
    if (idx >= B_ * D_) return;
    int d = idx & 63, b = idx >> 6;
    float s = 0.f;
    for (int n = 0; n < N_; ++n) {
        size_t o = ((size_t)(b * N_ + n)) * D_ + d;
        s += x1[o] + m[o] + Qc[o];
    }
    y[idx] = s * (1.0f / N_);
}

// ---------------------------------------------------------------------------
// head: out[b] = gelu(y @ Wh1 + bh1) @ Wh2 + bh2
// ---------------------------------------------------------------------------
__global__ __launch_bounds__(128)
void head_kernel(const float* __restrict__ y, const float* __restrict__ Wh1,
                 const float* __restrict__ bh1, const float* __restrict__ Wh2,
                 const float* __restrict__ bh2, float* __restrict__ out) {
    int b = threadIdx.x;
    if (b >= B_) return;
    const float* yr = y + (size_t)b * D_;
    float acc = bh2[0];
    for (int j = 0; j < 32; ++j) {
        float z = bh1[j];
        for (int d = 0; d < D_; ++d) z += yr[d] * Wh1[d * 32 + j];
        acc += gelu_f(z) * Wh2[j];
    }
    out[b] = acc;
}

// ---------------------------------------------------------------------------
extern "C" void kernel_launch(void* const* d_in, const int* in_sizes, int n_in,
                              void* d_out, int out_size, void* d_ws, size_t ws_size,
                              hipStream_t stream) {
    (void)in_sizes; (void)n_in; (void)out_size; (void)ws_size;
    const float* x      = (const float*)d_in[0];
    const float* W_emb  = (const float*)d_in[1];
    const float* b_emb  = (const float*)d_in[2];
    const float* rms_s  = (const float*)d_in[3];
    const float* Wq     = (const float*)d_in[4];
    const float* Wk     = (const float*)d_in[5];
    const float* Wv     = (const float*)d_in[6];
    const float* k_pos  = (const float*)d_in[7];
    const float* v_pos  = (const float*)d_in[8];
    const float* Wck    = (const float*)d_in[9];
    const float* bck    = (const float*)d_in[10];
    const float* Wcv    = (const float*)d_in[11];
    const float* bcv    = (const float*)d_in[12];
    const float* mem_ck = (const float*)d_in[13];
    const float* mem_cv = (const float*)d_in[14];
    const float* Wg     = (const float*)d_in[15];
    const float* bg     = (const float*)d_in[16];
    const float* Wo     = (const float*)d_in[17];
    const float* ln1_g  = (const float*)d_in[18];
    const float* ln1_b  = (const float*)d_in[19];
    const float* Wt1    = (const float*)d_in[20];
    const float* bt1    = (const float*)d_in[21];
    const float* Wt2    = (const float*)d_in[22];
    const float* bt2    = (const float*)d_in[23];
    const float* ln2_g  = (const float*)d_in[24];
    const float* ln2_b  = (const float*)d_in[25];
    const float* Wc1    = (const float*)d_in[26];
    const float* bc1    = (const float*)d_in[27];
    const float* Wc2    = (const float*)d_in[28];
    const float* bc2    = (const float*)d_in[29];
    const float* Wh1    = (const float*)d_in[30];
    const float* bh1    = (const float*)d_in[31];
    const float* Wh2    = (const float*)d_in[32];
    const float* bh2    = (const float*)d_in[33];
    float* out = (float*)d_out;

    char* ws = (char*)d_ws;
    size_t off = 0;
    auto alloc = [&](size_t bytes) -> void* {
        void* p = ws + off;
        off += (bytes + 255) & ~(size_t)255;
        return p;
    };
    constexpr size_t MN = (size_t)B_ * N_;              // 16384 rows
    float*    e      = (float*)   alloc(MN * D_ * 4);
    float*    hbuf   = (float*)   alloc(MN * D_ * 4);
    _Float16* h16    = (_Float16*)alloc(MN * D_ * 2);
    _Float16* t16    = (_Float16*)alloc((size_t)B_ * D_ * N_ * 2);
    float*    G      = (float*)   alloc(MN * 24 * 4);
    _Float16* attn16 = (_Float16*)alloc(MN * HD_ * 2);
    float*    x1     = (float*)   alloc(MN * D_ * 4);
    float*    ckb    = (float*)   alloc((size_t)B_ * H_ * NB_ * DH_ * 4);
    float*    cvb    = (float*)   alloc((size_t)B_ * H_ * NB_ * DH_ * 4);
    float*    mbuf   = (float*)   alloc(MN * D_ * 4);
    float*    Qc     = (float*)   alloc(MN * D_ * 4);
    float*    ybuf   = (float*)   alloc((size_t)B_ * D_ * 4);
    _Float16* Wq16   = (_Float16*)alloc(D_ * HD_ * 2);
    _Float16* Wk16   = (_Float16*)alloc(D_ * HD_ * 2);
    _Float16* Wv16   = (_Float16*)alloc(D_ * HD_ * 2);
    _Float16* Wo16   = (_Float16*)alloc(HD_ * D_ * 2);
    _Float16* Wck16  = (_Float16*)alloc(256 * D_ * 2);
    _Float16* Wcv16  = (_Float16*)alloc(256 * D_ * 2);
    _Float16* Wt116  = (_Float16*)alloc(N_ * 256 * 2);
    _Float16* Wt216  = (_Float16*)alloc(256 * N_ * 2);
    _Float16* Wc116  = (_Float16*)alloc(D_ * 256 * 2);
    _Float16* Wc216  = (_Float16*)alloc(256 * D_ * 2);
    float*    Qbuf   = (float*)   alloc(MN * HD_ * 4);  // region A (reused)
    float*    Kbuf   = (float*)   alloc(MN * HD_ * 4);  // region B (reused)
    float*    Vbuf   = (float*)   alloc(MN * HD_ * 4);
    _Float16* kb16   = (_Float16*)alloc((size_t)B_ * H_ * NB_ * 256 * 2);
    _Float16* vb16   = (_Float16*)alloc((size_t)B_ * H_ * NB_ * 256 * 2);
    // late-phase aliases (dead Q/K regions after attention)
    float*    U      = (float*)Qbuf;                          // 8192x256 f32
    _Float16* U16    = (_Float16*)((char*)Qbuf + 8388608);    // 8192x256 f16
    float*    Vt     = (float*)((char*)Qbuf + 12582912);      // 8192x128 f32
    float*    P      = (float*)Kbuf;                          // 16384x256 f32
    _Float16* P16    = (_Float16*)((char*)Kbuf + 16777216);   // 16384x256 f16
    _Float16* m2_16  = (_Float16*)((char*)Kbuf + 25165824);   // 16384x64 f16

    auto cvt = [&](const float* s, _Float16* d, int n) {
        cvt16_kernel<<<(n + 255) / 256, 256, 0, stream>>>(s, d, n);
    };
    cvt(Wq, Wq16, D_ * HD_);   cvt(Wk, Wk16, D_ * HD_);   cvt(Wv, Wv16, D_ * HD_);
    cvt(Wo, Wo16, HD_ * D_);   cvt(Wck, Wck16, 256 * D_); cvt(Wcv, Wcv16, 256 * D_);
    cvt(Wt1, Wt116, N_ * 256); cvt(Wt2, Wt216, 256 * N_);
    cvt(Wc1, Wc116, D_ * 256); cvt(Wc2, Wc216, 256 * D_);

    embed_kernel<<<(int)MN, 32, 0, stream>>>(x, W_emb, b_emb, rms_s, ln1_g, ln1_b,
                                             e, hbuf, h16, t16);

    // waves cover 16x64 output strips
    auto gblocks = [](int M, int Nc) { return ((M / 16) * (Nc / 64) + 7) / 8; };
    // QKV projections (no bias)
    gemm_wmma_kernel<0, false><<<gblocks(MN, HD_), 256, 0, stream>>>(
        h16, Wq16, nullptr, Qbuf, nullptr, (int)MN, HD_, D_);
    gemm_wmma_kernel<0, false><<<gblocks(MN, HD_), 256, 0, stream>>>(
        h16, Wk16, nullptr, Kbuf, nullptr, (int)MN, HD_, D_);
    gemm_wmma_kernel<0, false><<<gblocks(MN, HD_), 256, 0, stream>>>(
        h16, Wv16, nullptr, Vbuf, nullptr, (int)MN, HD_, D_);

    gate_kernel<<<(B_ * N_ * 24 + 255) / 256, 256, 0, stream>>>(hbuf, Wg, bg, G);

    kbvb_kernel<<<(B_ * H_ * NB_ * 256 + 255) / 256, 256, 0, stream>>>(
        Kbuf, Vbuf, k_pos, v_pos, kb16, vb16);

    int Mck = B_ * H_ * NB_;  // 32768
    gemm_wmma_kernel<0, false><<<gblocks(Mck, D_), 256, 0, stream>>>(
        kb16, Wck16, bck, ckb, nullptr, Mck, D_, 256);
    gemm_wmma_kernel<0, false><<<gblocks(Mck, D_), 256, 0, stream>>>(
        vb16, Wcv16, bcv, cvb, nullptr, Mck, D_, 256);

    attn_kernel<<<B_ * H_ * N_ / 4, 128, 0, stream>>>(
        Qbuf, Kbuf, Vbuf, ckb, cvb, mem_ck, mem_cv, G, attn16);

    // x1 = attn @ Wo  (Qbuf/Kbuf now dead -> aliased below)
    gemm_wmma_kernel<0, false><<<gblocks(MN, D_), 256, 0, stream>>>(
        attn16, Wo16, nullptr, x1, nullptr, (int)MN, D_, HD_);

    // token mixer: (B*D, N) @ (N,256) -> gelu -> @ (256,N) + bt2
    int Mt = B_ * D_;  // 8192
    gemm_wmma_kernel<1, true><<<gblocks(Mt, 256), 256, 0, stream>>>(
        t16, Wt116, bt1, U, U16, Mt, 256, N_);
    gemm_wmma_kernel<0, false><<<gblocks(Mt, N_), 256, 0, stream>>>(
        U16, Wt216, bt2, Vt, nullptr, Mt, N_, 256);
    addT_kernel<<<(B_ * N_ * D_ + 255) / 256, 256, 0, stream>>>(e, Vt, mbuf);

    // channel mixer
    ln_kernel<<<(int)MN, 32, 0, stream>>>(mbuf, ln2_g, ln2_b, m2_16);
    gemm_wmma_kernel<1, true><<<gblocks(MN, 256), 256, 0, stream>>>(
        m2_16, Wc116, bc1, P, P16, (int)MN, 256, D_);
    gemm_wmma_kernel<0, false><<<gblocks(MN, D_), 256, 0, stream>>>(
        P16, Wc216, bc2, Qc, nullptr, (int)MN, D_, 256);

    mean_kernel<<<(B_ * D_ + 255) / 256, 256, 0, stream>>>(x1, mbuf, Qc, ybuf);
    head_kernel<<<1, 128, 0, stream>>>(ybuf, Wh1, bh1, Wh2, bh2, out);
}
